// WindowAttention_84576495993028
// MI455X (gfx1250) — compile-verified, matrix operand order checked by reading, not run on hardware
//
#include <hip/hip_runtime.h>

typedef __attribute__((ext_vector_type(8)))  __bf16 bf16x8;
typedef __attribute__((ext_vector_type(16))) __bf16 bf16x16;
typedef __attribute__((ext_vector_type(8)))  float  f32x8;

#define NTOK   131072        // 8*128*128 tokens
#define NWIN   12288         // 8*16*16 windows * 6 heads
#define QSCALE 0.17677669529663687f   // 1/sqrt(32)

__device__ __forceinline__ __bf16 to_bf16(float f) {
  union { float f; unsigned u; } v; v.f = f;
  unsigned r = v.u + 0x7FFFu + ((v.u >> 16) & 1u);   // round-to-nearest-even
  union { unsigned short s; __bf16 b; } o; o.s = (unsigned short)(r >> 16);
  return o.b;
}
__device__ __forceinline__ float bf2f(__bf16 b) {
  union { __bf16 b; unsigned short s; } i; i.b = b;
  union { unsigned u; float f; } o; o.u = ((unsigned)i.s) << 16;
  return o.f;
}

// Load one 16-lane WMMA fragment (A or B operand) from a row-major bf16 matrix.
// ISA 16-bit operand layout: lane l (g=l>>4, r=l&15) holds row r, and
// K-values {k0+8g .. k0+8g+7} then {k0+16+8g .. k0+16+8g+7}: two b128 loads.
__device__ __forceinline__ bf16x16 load_frag(const __bf16* base, int row, int ld,
                                             int k0, int g) {
  const __bf16* p = base + (size_t)row * ld + k0 + 8 * g;
  bf16x8 lo = *(const bf16x8*)(p);
  bf16x8 hi = *(const bf16x8*)(p + 16);
  return __builtin_shufflevector(lo, hi, 0,1,2,3,4,5,6,7,8,9,10,11,12,13,14,15);
}

// ---------------- K0: conversions ----------------
__global__ __launch_bounds__(256) void k_convert_x(const float* __restrict__ x,
                                                   __bf16* __restrict__ xbf, int n) {
  int i = blockIdx.x * blockDim.x + threadIdx.x;
  if (i < n) xbf[i] = to_bf16(x[i]);
}

// w: K x N row-major (fp32)  ->  wT: N x K row-major (bf16)
__global__ __launch_bounds__(256) void k_transpose_w(const float* __restrict__ w,
                                                     __bf16* __restrict__ wT,
                                                     int K, int N) {
  int i = blockIdx.x * blockDim.x + threadIdx.x;
  if (i < K * N) { int k = i / N, n = i - k * N; wT[n * K + k] = to_bf16(w[i]); }
}

// ---------------- K1: QKV GEMM + window partition ----------------
// grid (512, 9), block 256 (8 waves).  Wave tile: 32 M x 64 N, K = 192.
// 8 WMMAs per k-step against 12 b128 loads (2 A frags share each B frag).
__global__ __launch_bounds__(256) void k_qkv_gemm(const __bf16* __restrict__ xbf,
                                                  const __bf16* __restrict__ wT,
                                                  const float*  __restrict__ bias,
                                                  __bf16* __restrict__ qbuf,
                                                  __bf16* __restrict__ kbuf,
                                                  __bf16* __restrict__ vtbuf) {
  const int lane = threadIdx.x & 31, wv = threadIdx.x >> 5;
  const int g = lane >> 4, mr = lane & 15;
  const int row0 = blockIdx.x * 256 + wv * 32;
  const int n0   = blockIdx.y * 64;

  f32x8 acc[2][4] = {};
  for (int kk = 0; kk < 192; kk += 32) {
    bf16x16 a0 = load_frag(xbf, row0 + mr,      192, kk, g);
    bf16x16 a1 = load_frag(xbf, row0 + 16 + mr, 192, kk, g);
#pragma unroll
    for (int t = 0; t < 4; ++t) {
      bf16x16 b = load_frag(wT, n0 + t * 16 + mr, 192, kk, g);
      acc[0][t] = __builtin_amdgcn_wmma_f32_16x16x32_bf16(false, a0, false, b,
                                                          (short)0, acc[0][t], false, false);
      acc[1][t] = __builtin_amdgcn_wmma_f32_16x16x32_bf16(false, a1, false, b,
                                                          (short)0, acc[1][t], false, false);
    }
  }
  // Epilogue: bias, split into q/k/v, window-partition. col = s*192 + head*32 + n
#pragma unroll
  for (int t = 0; t < 4; ++t) {
    const int colb   = n0 + t * 16;
    const int s      = colb / 192;            // uniform per tile
    const int within = colb - s * 192;
    const int head   = within >> 5;
    const int nb     = within & 31;           // 0 or 16
    const int n      = nb + mr;
    const float bv   = bias[colb + mr];
#pragma unroll
    for (int rb = 0; rb < 2; ++rb) {
#pragma unroll
      for (int r = 0; r < 8; ++r) {
        const int m   = row0 + rb * 16 + r + 8 * g;   // token index
        const int b_  = m >> 14;                      // /(128*128)
        const int rem = m & 16383;
        const int h = rem >> 7, w = rem & 127;
        const int wy = h >> 3, iy = h & 7, wx = w >> 3, ix = w & 7;
        const int ti = ((b_ * 16 + wy) * 16 + wx) * 6 + head;
        const int ii = iy * 8 + ix;                   // token-in-window
        const float v = acc[rb][t][r] + bv;
        if (s == 0)      qbuf [ti * 2048 + ii * 32 + n] = to_bf16(v * QSCALE);
        else if (s == 1) kbuf [ti * 2048 + ii * 32 + n] = to_bf16(v);
        else             vtbuf[ti * 2048 + n * 64 + ii] = to_bf16(v);  // V^T: [hd][64]
      }
    }
  }
}

// ---------------- K2: windowed attention ----------------
// one block (4 waves) per (window, head); q/k: [64][32], v^T: [32][64]
__global__ __launch_bounds__(128) void k_attn(const __bf16* __restrict__ qbuf,
                                              const __bf16* __restrict__ kbuf,
                                              const __bf16* __restrict__ vtbuf,
                                              __bf16* __restrict__ obuf) {
  __shared__ __bf16 Plds[64 * 64];
  const int t = blockIdx.x;
  const int lane = threadIdx.x & 31, wv = threadIdx.x >> 5;
  const int g = lane >> 4, mr = lane & 15;
  const __bf16* qb = qbuf + (size_t)t * 2048;
  const __bf16* kb = kbuf + (size_t)t * 2048;
  const __bf16* vb = vtbuf + (size_t)t * 2048;

  // S = q k^T  : wave handles 16 rows x 64 cols, contraction hd=32 (one WMMA/tile)
  f32x8 s[4];
  bf16x16 a = load_frag(qb, wv * 16 + mr, 32, 0, g);
#pragma unroll
  for (int jt = 0; jt < 4; ++jt) {
    f32x8 z = {};
    bf16x16 b = load_frag(kb, jt * 16 + mr, 32, 0, g);
    s[jt] = __builtin_amdgcn_wmma_f32_16x16x32_bf16(false, a, false, b,
                                                    (short)0, z, false, false);
  }
  // Row softmax: C-layout means lane holds col (l&15), rows r+8g -> reduce over
  // lanes 0..15 within each half via xor shuffles (masks < 16 preserve halves).
#pragma unroll
  for (int r = 0; r < 8; ++r) {
    float mx = fmaxf(fmaxf(s[0][r], s[1][r]), fmaxf(s[2][r], s[3][r]));
#pragma unroll
    for (int off = 1; off < 16; off <<= 1) mx = fmaxf(mx, __shfl_xor(mx, off, 32));
    float sum = 0.f;
#pragma unroll
    for (int jt = 0; jt < 4; ++jt) {
      float p = __expf(s[jt][r] - mx);
      s[jt][r] = p; sum += p;
    }
#pragma unroll
    for (int off = 1; off < 16; off <<= 1) sum += __shfl_xor(sum, off, 32);
    const float inv = 1.f / sum;
    const int row = wv * 16 + r + 8 * g;
#pragma unroll
    for (int jt = 0; jt < 4; ++jt)
      Plds[row * 64 + jt * 16 + mr] = to_bf16(s[jt][r] * inv);
  }
  __syncthreads();   // redistribute P rows -> lanes for the A-operand

  // O = P V : 16 rows x 32 cols per wave, contraction 64 (2 WMMAs per n-tile)
  const int head = t % 6, wid = t / 6;
  const int wx = wid & 15, wy = (wid >> 4) & 15, b_ = wid >> 8;
#pragma unroll
  for (int nt = 0; nt < 2; ++nt) {
    f32x8 oa = {};
#pragma unroll
    for (int kk = 0; kk < 2; ++kk) {
      bf16x16 pa = load_frag((const __bf16*)Plds, wv * 16 + mr, 64, kk * 32, g);
      bf16x16 vf = load_frag(vb, nt * 16 + mr, 64, kk * 32, g);  // B[j][n] = v^T[n][j]
      oa = __builtin_amdgcn_wmma_f32_16x16x32_bf16(false, pa, false, vf,
                                                   (short)0, oa, false, false);
    }
    // window reverse -> [B,H,W,C] (bf16 for dwconv input)
#pragma unroll
    for (int r = 0; r < 8; ++r) {
      const int ii = wv * 16 + r + 8 * g;
      const int iy = ii >> 3, ix = ii & 7;
      const int h = wy * 8 + iy, w = wx * 8 + ix;
      const int tok = (b_ * 128 + h) * 128 + w;
      obuf[(size_t)tok * 192 + head * 32 + nt * 16 + mr] = to_bf16(oa[r]);
    }
  }
}

// ---------------- K3: depthwise 3x3 conv ----------------
// one thread per (token, 8-channel group): 9 b128 loads, fp32 accumulate
__global__ __launch_bounds__(256) void k_dwconv(const __bf16* __restrict__ obuf,
                                                const float* __restrict__ wdw,
                                                const float* __restrict__ bdw,
                                                __bf16* __restrict__ dbuf) {
  const int tid = blockIdx.x * blockDim.x + threadIdx.x;
  const int grp = tid % 24, tok = tid / 24;
  if (tok >= NTOK) return;
  const int c0 = grp * 8;
  const int rem = tok & 16383;
  const int h = rem >> 7, w = rem & 127;
  float accf[8];
#pragma unroll
  for (int j = 0; j < 8; ++j) accf[j] = bdw[c0 + j];
#pragma unroll
  for (int dy = -1; dy <= 1; ++dy) {
#pragma unroll
    for (int dx = -1; dx <= 1; ++dx) {
      const int hh = h + dy, ww = w + dx;
      if (hh < 0 || hh > 127 || ww < 0 || ww > 127) continue;
      bf16x8 vv = *(const bf16x8*)(obuf + (size_t)(tok + dy * 128 + dx) * 192 + c0);
      const float* wk = wdw + ((dy + 1) * 3 + (dx + 1)) * 192 + c0;
#pragma unroll
      for (int j = 0; j < 8; ++j) accf[j] += bf2f(vv[j]) * wk[j];
    }
  }
  bf16x8 outv;
#pragma unroll
  for (int j = 0; j < 8; ++j) outv[j] = to_bf16(accf[j]);
  *(bf16x8*)(dbuf + (size_t)tok * 192 + c0) = outv;
}

// ---------------- K4: projection GEMM (fp32 out) ----------------
// grid (512, 3), block 256 (8 waves).  Wave tile: 32 M x 64 N, K = 192.
__global__ __launch_bounds__(256) void k_proj_gemm(const __bf16* __restrict__ dbuf,
                                                   const __bf16* __restrict__ wT,
                                                   const float* __restrict__ bias,
                                                   float* __restrict__ out) {
  const int lane = threadIdx.x & 31, wv = threadIdx.x >> 5;
  const int g = lane >> 4, mr = lane & 15;
  const int row0 = blockIdx.x * 256 + wv * 32;
  const int n0   = blockIdx.y * 64;
  f32x8 acc[2][4] = {};
  for (int kk = 0; kk < 192; kk += 32) {
    bf16x16 a0 = load_frag(dbuf, row0 + mr,      192, kk, g);
    bf16x16 a1 = load_frag(dbuf, row0 + 16 + mr, 192, kk, g);
#pragma unroll
    for (int t = 0; t < 4; ++t) {
      bf16x16 b = load_frag(wT, n0 + t * 16 + mr, 192, kk, g);
      acc[0][t] = __builtin_amdgcn_wmma_f32_16x16x32_bf16(false, a0, false, b,
                                                          (short)0, acc[0][t], false, false);
      acc[1][t] = __builtin_amdgcn_wmma_f32_16x16x32_bf16(false, a1, false, b,
                                                          (short)0, acc[1][t], false, false);
    }
  }
#pragma unroll
  for (int t = 0; t < 4; ++t) {
    const int col = n0 + t * 16 + mr;
    const float bv = bias[col];
#pragma unroll
    for (int rb = 0; rb < 2; ++rb)
#pragma unroll
      for (int r = 0; r < 8; ++r)
        out[(size_t)(row0 + rb * 16 + r + 8 * g) * 192 + col] = acc[rb][t][r] + bv;
  }
}

extern "C" void kernel_launch(void* const* d_in, const int* in_sizes, int n_in,
                              void* d_out, int out_size, void* d_ws, size_t ws_size,
                              hipStream_t stream) {
  const float* x      = (const float*)d_in[0];
  const float* w_qkv  = (const float*)d_in[1];
  const float* b_qkv  = (const float*)d_in[2];
  const float* w_dw   = (const float*)d_in[3];
  const float* b_dw   = (const float*)d_in[4];
  const float* w_proj = (const float*)d_in[5];
  const float* b_proj = (const float*)d_in[6];
  (void)in_sizes; (void)n_in; (void)out_size; (void)ws_size;

  char* ws = (char*)d_ws;
  size_t off = 0;
  auto alloc = [&](size_t bytes) -> char* {
    char* p = ws + off;
    off += bytes; off = (off + 255) & ~(size_t)255;
    return p;
  };
  __bf16* xbf    = (__bf16*)alloc((size_t)NTOK * 192 * 2);   // also reused as dwconv out
  __bf16* wqkvT  = (__bf16*)alloc((size_t)576 * 192 * 2);
  __bf16* wprojT = (__bf16*)alloc((size_t)192 * 192 * 2);
  __bf16* qbuf   = (__bf16*)alloc((size_t)NWIN * 2048 * 2);
  __bf16* kbuf   = (__bf16*)alloc((size_t)NWIN * 2048 * 2);
  __bf16* vtbuf  = (__bf16*)alloc((size_t)NWIN * 2048 * 2);
  __bf16* obuf   = (__bf16*)alloc((size_t)NTOK * 192 * 2);
  __bf16* dbuf   = xbf;   // xbf dead after K1; reuse for dwconv output

  k_convert_x <<<(NTOK * 192 + 255) / 256, 256, 0, stream>>>(x, xbf, NTOK * 192);
  k_transpose_w<<<(192 * 576 + 255) / 256, 256, 0, stream>>>(w_qkv, wqkvT, 192, 576);
  k_transpose_w<<<(192 * 192 + 255) / 256, 256, 0, stream>>>(w_proj, wprojT, 192, 192);
  k_qkv_gemm  <<<dim3(512, 9), 256, 0, stream>>>(xbf, wqkvT, b_qkv, qbuf, kbuf, vtbuf);
  k_attn      <<<NWIN, 128, 0, stream>>>(qbuf, kbuf, vtbuf, obuf);
  k_dwconv    <<<(NTOK * 24) / 256, 256, 0, stream>>>(obuf, w_dw, b_dw, dbuf);
  k_proj_gemm <<<dim3(512, 3), 256, 0, stream>>>(dbuf, wprojT, b_proj, (float*)d_out);
}
